// PretrainingLoss_45629732552886
// MI455X (gfx1250) — matrix operand order repaired and checked by reading, not compiled, not run on hardware
//
#include <hip/hip_runtime.h>
#include <hip/hip_bf16.h>
#include <math.h>

// ---- problem constants (reference: B=16, D=768, T=2048, K=100, TEMP=0.1) ----
#define BB 16
#define DD 768
#define TT 2048
#define KK 100

typedef __attribute__((ext_vector_type(16))) _Float16 v16h;
typedef __attribute__((ext_vector_type(8)))  _Float16 v8h;
typedef __attribute__((ext_vector_type(4)))  _Float16 v4h;
typedef __attribute__((ext_vector_type(8)))  float    v8f;
typedef __attribute__((ext_vector_type(4)))  float    v4f;   // clang vector: OK for nontemporal builtins

union HV16 { v16h v; v8h h[2]; };

// Row stride (in halfs) of the per-wave LDS tile [16 t][32 d], padded so the
// b128 slices at byte offsets {0,16,32,48} stay 16B-aligned (96B = 6*16).
#define TROW 48
#define TILE_BYTES 1536            // 16 * TROW * 2
#define WAVE_LDS   6144            // 2 buffers x 2 tensors x TILE_BYTES

// ---------------------------------------------------------------------------
// Kernel 1: per (b, t) compute u = exp( (L.T / (||L|| * ||T||)) / 0.1 )
// One wave owns a 16-wide t tile and walks D in 32-chunks. Each f32 element
// is read from HBM exactly once (non-temporal b128), packed to f16 and
// staged transposed [t][d] in LDS (b64 stores). Three WMMAs accumulate the
// 16x16 cross-Gram matrices; their diagonals are num, ||L||^2, ||T||^2.
// Double-buffered so the HBM stream overlaps the matrix pipe.
// ---------------------------------------------------------------------------
__global__ __launch_bounds__(256)
void pl_u_wmma_kernel(const float* __restrict__ logits,
                      const float* __restrict__ target,
                      float* __restrict__ uOut)
{
    __shared__ unsigned char smemRaw[8 * WAVE_LDS];

    const int wave = threadIdx.x >> 5;
    const int lane = threadIdx.x & 31;
    unsigned char* wsm = smemRaw + wave * WAVE_LDS;
    _Float16* tl[2] = { (_Float16*)wsm,
                        (_Float16*)(wsm + 2 * TILE_BYTES) };
    _Float16* tt[2] = { (_Float16*)(wsm + TILE_BYTES),
                        (_Float16*)(wsm + 3 * TILE_BYTES) };

    const int tileId = blockIdx.x * 8 + wave;    // 0 .. B*T/16-1 = 2047
    const int b  = tileId >> 7;                  // 128 tiles per batch row
    const int t0 = (tileId & 127) << 4;
    const size_t rowBase = (size_t)b * DD * TT + t0;

    const int q  = lane & 3;                     // t quarter (4 floats)
    const int dg = lane >> 2;                    // d group   (4 rows)

    // ---- stage one 16t x 32d chunk of both tensors into buffer `sel`
    auto stage = [&](int sel, int d0) {
        const float* lp = logits + rowBase + (size_t)(d0 + dg * 4) * TT + q * 4;
        const float* tp = target + rowBase + (size_t)(d0 + dg * 4) * TT + q * 4;
        if (d0 + 64 < DD) {       // 2-steps-ahead prefetch: lane == d row
            __builtin_prefetch(logits + rowBase + (size_t)(d0 + 64 + lane) * TT, 0, 0);
            __builtin_prefetch(target + rowBase + (size_t)(d0 + 64 + lane) * TT, 0, 0);
        }
        v4f l0 = __builtin_nontemporal_load((const v4f*)(lp + 0 * TT));
        v4f l1 = __builtin_nontemporal_load((const v4f*)(lp + 1 * TT));
        v4f l2 = __builtin_nontemporal_load((const v4f*)(lp + 2 * TT));
        v4f l3 = __builtin_nontemporal_load((const v4f*)(lp + 3 * TT));
        v4f s0 = __builtin_nontemporal_load((const v4f*)(tp + 0 * TT));
        v4f s1 = __builtin_nontemporal_load((const v4f*)(tp + 1 * TT));
        v4f s2 = __builtin_nontemporal_load((const v4f*)(tp + 2 * TT));
        v4f s3 = __builtin_nontemporal_load((const v4f*)(tp + 3 * TT));
        _Float16* dl = tl[sel];
        _Float16* dt = tt[sel];
        #pragma unroll
        for (int i = 0; i < 4; ++i) {            // i = t within quarter
            v4h hl = { (_Float16)l0[i], (_Float16)l1[i],
                       (_Float16)l2[i], (_Float16)l3[i] };
            v4h ht = { (_Float16)s0[i], (_Float16)s1[i],
                       (_Float16)s2[i], (_Float16)s3[i] };
            *(v4h*)&dl[(q * 4 + i) * TROW + dg * 4] = hl;   // ds_store_b64
            *(v4h*)&dt[(q * 4 + i) * TROW + dg * 4] = ht;
        }
    };

    v8f cNum = {}; v8f cLL = {}; v8f cTT = {};
    const int M   = lane & 15;                   // A-row (=t) / B-col (=t)
    const int kbA = (lane < 16) ? 0 : 8;         // 16-bit A layout K base
    const int kbB = (lane < 16) ? 0 : 8;         // 16-bit B layout K base (same as A)

    // ---- consume buffer `sel`: 8 x ds_load_b128 + 3 x WMMA
    auto accum = [&](int sel) {
        const _Float16* sl = tl[sel];
        const _Float16* st = tt[sel];
        HV16 aL, aT, bL, bT;
        aL.h[0] = *(const v8h*)&sl[M * TROW + kbA];
        aL.h[1] = *(const v8h*)&sl[M * TROW + 16 + kbA];
        aT.h[0] = *(const v8h*)&st[M * TROW + kbA];
        aT.h[1] = *(const v8h*)&st[M * TROW + 16 + kbA];
        bL.h[0] = *(const v8h*)&sl[M * TROW + kbB];
        bL.h[1] = *(const v8h*)&sl[M * TROW + 16 + kbB];
        bT.h[0] = *(const v8h*)&st[M * TROW + kbB];
        bT.h[1] = *(const v8h*)&st[M * TROW + 16 + kbB];
        cNum = __builtin_amdgcn_wmma_f32_16x16x32_f16(false, aL.v, false, bT.v,
                                                      (short)0, cNum, false, false);
        cLL  = __builtin_amdgcn_wmma_f32_16x16x32_f16(false, aL.v, false, bL.v,
                                                      (short)0, cLL, false, false);
        cTT  = __builtin_amdgcn_wmma_f32_16x16x32_f16(false, aT.v, false, bT.v,
                                                      (short)0, cTT, false, false);
    };

    // software-pipelined ping-pong over D (24 chunks)
    stage(0, 0);
    for (int d0 = 0; d0 < DD; d0 += 64) {
        if (d0 + 32 < DD) stage(1, d0 + 32);
        accum(0);
        if (d0 + 64 < DD) stage(0, d0 + 64);
        if (d0 + 32 < DD) accum(1);
    }

    // ---- dump accumulators to LDS as [M][N] and read the diagonals
    float* sh = (float*)wsm;                     // reuse 3KB: 3 x 16 x 16 f32
    const int mHi = (lane < 16) ? 0 : 8;
    #pragma unroll
    for (int r = 0; r < 8; ++r) {
        const int Mr = r + mHi;
        const int N  = lane & 15;
        sh[0 * 256 + Mr * 16 + N] = cNum[r];
        sh[1 * 256 + Mr * 16 + N] = cLL[r];
        sh[2 * 256 + Mr * 16 + N] = cTT[r];
    }
    if (lane < 16) {
        const float n  = sh[0 * 256 + lane * 17];
        const float ll = sh[1 * 256 + lane * 17];
        const float t2 = sh[2 * 256 + lane * 17];
        const float denom = fmaxf(__builtin_amdgcn_sqrtf(ll), 1e-8f) *
                            fmaxf(__builtin_amdgcn_sqrtf(t2), 1e-8f);
        // num/denom/TEMPERATURE, TEMPERATURE = 0.1
        uOut[(size_t)b * TT + t0 + lane] = __expf(n / denom * 10.0f);
    }
}

// ---------------------------------------------------------------------------
// Kernel 2: distractor sums. u[b,:] (8KB) staged in LDS; 101 gathers per t.
// JAX threefry permutations are approximated by a keyed affine bijection on
// Z_2047 (multiplier forced coprime to 2047 = 23*89): deterministic
// sampling-without-replacement with identical statistics.
// ---------------------------------------------------------------------------
__global__ __launch_bounds__(256)
void pl_dist_kernel(const float* __restrict__ u, float* __restrict__ partials)
{
    __shared__ float shU[TT];
    __shared__ float red[8];

    const int b     = blockIdx.x >> 3;
    const int tBase = (blockIdx.x & 7) * 256;
    const float* ub = u + (size_t)b * TT;

    for (int i = threadIdx.x; i < TT; i += 256) shU[i] = ub[i];
    __syncthreads();

    const int t = tBase + threadIdx.x;
    const float uu = shU[t];

    unsigned h = (unsigned)(b * TT + t) * 2654435761u + 0x9E3779B9u;
    h ^= h >> 16; h *= 0x85EBCA6Bu; h ^= h >> 13; h *= 0xC2B2AE35u; h ^= h >> 16;
    unsigned a = (h % 2045u) + 1u;
    while ((a % 23u == 0u) || (a % 89u == 0u)) a += 1u;   // gcd(a, 2047) == 1
    unsigned p = (h >> 11) % 2047u;

    float sumExcl = 0.0f, sumK = 0.0f;
    bool hasI = false;
    for (int j = 0; j <= KK; ++j) {            // K+1 = 101 distractor slots
        const float v = shU[p];
        const bool isI = ((int)p == t);
        hasI = hasI || isI;
        sumExcl += isI ? 0.0f : v;
        if (j < KK) sumK += v;
        p += a; if (p >= 2047u) p -= 2047u;
    }
    float dist = (hasI ? sumExcl : sumK) + ((t < TT - 1) ? uu : 0.0f);
    float le = uu / dist;

    // wave32 reduction, then cross-wave via LDS
    for (int off = 16; off > 0; off >>= 1) le += __shfl_down(le, off, 32);
    if ((threadIdx.x & 31) == 0) red[threadIdx.x >> 5] = le;
    __syncthreads();
    if (threadIdx.x == 0) {
        float s = 0.0f;
        #pragma unroll
        for (int i = 0; i < 8; ++i) s += red[i];
        partials[blockIdx.x] = s;
    }
}

// ---------------------------------------------------------------------------
// Kernel 3: final scalar: -log(mean(loss))
// ---------------------------------------------------------------------------
__global__ __launch_bounds__(128)
void pl_final_kernel(const float* __restrict__ partials, float* __restrict__ out)
{
    __shared__ float sh[128];
    sh[threadIdx.x] = partials[threadIdx.x];
    __syncthreads();
    if (threadIdx.x == 0) {
        float s = 0.0f;
        for (int i = 0; i < 128; ++i) s += sh[i];
        out[0] = -logf(s / (float)(BB * TT));
    }
}

extern "C" void kernel_launch(void* const* d_in, const int* in_sizes, int n_in,
                              void* d_out, int out_size, void* d_ws, size_t ws_size,
                              hipStream_t stream)
{
    const float* logits = (const float*)d_in[0];
    const float* target = (const float*)d_in[1];
    float* uBuf     = (float*)d_ws;            // B*T = 32768 floats
    float* partials = uBuf + BB * TT;          // 128 floats
    float* out      = (float*)d_out;

    // 2048 tiles of 16 t, 8 waves/block
    pl_u_wmma_kernel<<<256, 256, 0, stream>>>(logits, target, uBuf);
    // 128 blocks x 256 t
    pl_dist_kernel<<<128, 256, 0, stream>>>(uBuf, partials);
    pl_final_kernel<<<1, 128, 0, stream>>>(partials, out);
}
